// MIFModule_42563125903832
// MI455X (gfx1250) — matrix operand order repaired and checked by reading, not compiled, not run on hardware
//
#include <hip/hip_runtime.h>
#include <hip/hip_bf16.h>

typedef __bf16 bf16_t;
typedef __attribute__((ext_vector_type(16))) __bf16 v16bf;
typedef __attribute__((ext_vector_type(8)))  __bf16 v8bf;
typedef __attribute__((ext_vector_type(8)))  float  v8f;

constexpr int Bc  = 4;
constexpr int Cc  = 192;
constexpr int Lc  = 4096;          // H*W (64x64)
constexpr int DIc = 384;           // EXPAND*C
constexpr int MT  = Bc * Lc;       // 16384 rows for all GEMMs

// ---------------------------------------------------------------------------
// CDNA5 async copy: global -> LDS through the async engine (ASYNCcnt).
// ---------------------------------------------------------------------------
__device__ __forceinline__ void async_b128(unsigned lds_off, const void* gptr) {
  asm volatile("global_load_async_to_lds_b128 %0, %1, off"
               :: "v"(lds_off), "v"((unsigned long long)gptr) : "memory");
}

// ---------------------------------------------------------------------------
// Layout transforms / dtype conversion (one-time, tiny)
// ---------------------------------------------------------------------------
__global__ void k_cvt_x(const float* __restrict__ x, bf16_t* __restrict__ xbf) {
  int i = blockIdx.x * blockDim.x + threadIdx.x;     // B*L*C, c fastest
  if (i >= Bc * Cc * Lc) return;
  int c = i % Cc;
  int l = (i / Cc) % Lc;
  int b = i / (Cc * Lc);
  xbf[((size_t)(b * Lc + l)) * Cc + c] = (bf16_t)x[((size_t)(b * Cc + c)) * Lc + l];
}

__global__ void k_f32_to_bf16(const float* __restrict__ s, bf16_t* __restrict__ d, int n) {
  int i = blockIdx.x * blockDim.x + threadIdx.x;
  if (i < n) d[i] = (bf16_t)s[i];
}

// W_in (C,768) -> Bt (768,C)
__global__ void k_prep_WinT(const float* __restrict__ s, bf16_t* __restrict__ d) {
  int i = blockIdx.x * blockDim.x + threadIdx.x;     // 768*192, i = n*192 + c
  if (i >= 768 * Cc) return;
  int c = i % Cc, n = i / Cc;
  d[i] = (bf16_t)s[c * 768 + n];
}

// W_out (DI,C) -> Bt (C,DI)
__global__ void k_prep_WoutT(const float* __restrict__ s, bf16_t* __restrict__ d) {
  int i = blockIdx.x * blockDim.x + threadIdx.x;     // 192*384, i = c*384 + dd
  if (i >= Cc * DIc) return;
  int dd = i % DIc, c = i / DIc;
  d[i] = (bf16_t)s[dd * Cc + c];
}

// x_proj_w (K,44,DI) -> Bt (K, 64, DI)  (rows are r, zero padded 44->64)
__global__ void k_prep_xpw(const float* __restrict__ s, bf16_t* __restrict__ d) {
  int i = blockIdx.x * blockDim.x + threadIdx.x;     // K*64*DI
  if (i >= 4 * 64 * DIc) return;
  int dd = i % DIc;
  int r  = (i / DIc) % 64;
  int k  = i / (DIc * 64);
  float v = (r < 44) ? s[(k * 44 + r) * DIc + dd] : 0.f;
  d[i] = (bf16_t)v;
}

// dt_proj_w (K,DI,12) -> Bt (K, DI, 32)  (rows are d, K-dim zero padded 12->32)
__global__ void k_prep_dtw(const float* __restrict__ s, bf16_t* __restrict__ d) {
  int i = blockIdx.x * blockDim.x + threadIdx.x;     // K*DI*32
  if (i >= 4 * DIc * 32) return;
  int r  = i % 32;
  int dd = (i / 32) % DIc;
  int k  = i / (32 * DIc);
  float v = (r < 12) ? s[((k * DIc) + dd) * 12 + r] : 0.f;
  d[i] = (bf16_t)v;
}

// x_dbl (K,M,48) -> dtr bf16 padded (K,M,32)
__global__ void k_pack_dtr(const float* __restrict__ s, bf16_t* __restrict__ d) {
  int i = blockIdx.x * blockDim.x + threadIdx.x;     // K*M*32
  if (i >= 4 * MT * 32) return;
  int r = i & 31;
  size_t km = (size_t)(i >> 5);                      // k*M + m
  float v = (r < 12) ? s[km * 48 + r] : 0.f;
  d[i] = (bf16_t)v;
}

// ---------------------------------------------------------------------------
// bf16 WMMA GEMM:  C(MxN) = A(MxK) * Bt(NxK)^T   [+bias] [softplus->bf16]
//  * block tile 128x64, 8 waves, 2x2 v_wmma_f32_16x16x32_bf16 tiles per wave
//  * A and Bt staged by global_load_async_to_lds_b128, double buffered,
//    exactly 3 async B128 per thread per stage -> s_wait_asynccnt 3 pipelines
//  * M % 128 == 0, K % 32 == 0, Bt rows padded to a multiple of 64 (caller)
// ---------------------------------------------------------------------------
__global__ __launch_bounds__(256) void k_gemm_bf16(
    const bf16_t* __restrict__ A, const bf16_t* __restrict__ Bt,
    float* __restrict__ Cf, bf16_t* __restrict__ Cb,
    const float* __restrict__ bias, int M, int N, int Kd, int mode)
{
  __shared__ alignas(16) bf16_t As[2][128][40];   // 20.0 KB (row pitch 80B, 16B-aligned)
  __shared__ alignas(16) bf16_t Bs[2][64][40];    // 10.0 KB

  int tid  = threadIdx.x;
  int lane = tid & 31;
  int wave = tid >> 5;
  int wm   = wave & 3;          // 4 wave rows -> 128 M (32 each)
  int wn   = wave >> 2;         // 2 wave cols -> 64 N  (32 each)
  int half = lane >> 4;
  int ln15 = lane & 15;
  int bM   = blockIdx.y * 128;
  int bN   = blockIdx.x * 64;

  v8f acc[2][2];
#pragma unroll
  for (int mm = 0; mm < 2; ++mm)
#pragma unroll
    for (int nn = 0; nn < 2; ++nn)
#pragma unroll
      for (int r = 0; r < 8; ++r) acc[mm][nn][r] = 0.f;

  // stage issue: A 128 rows x 4 chunks (2/thread) + B 64 rows x 4 chunks (1/thread)
  auto stage_load = [&](int k0, int s) {
#pragma unroll
    for (int it = 0; it < 2; ++it) {
      int idx = tid + it * 256;                     // 0..511
      int r = idx >> 2, ch = idx & 3;
      async_b128((unsigned)(size_t)&As[s][r][ch * 8],
                 &A[(size_t)(bM + r) * Kd + k0 + ch * 8]);
    }
    {
      int r = tid >> 2, ch = tid & 3;
      async_b128((unsigned)(size_t)&Bs[s][r][ch * 8],
                 &Bt[(size_t)(bN + r) * Kd + k0 + ch * 8]);
    }
  };

  int nk = Kd >> 5;
  stage_load(0, 0);

  for (int s = 0; s < nk; ++s) {
    int cur = s & 1;
    if (s + 1 < nk) {
      stage_load((s + 1) * 32, cur ^ 1);
      asm volatile("s_wait_asynccnt 0x3" ::: "memory");   // stage s complete
    } else {
      asm volatile("s_wait_asynccnt 0x0" ::: "memory");
    }
    __syncthreads();

    const bf16_t (*cA)[40] = As[cur];
    const bf16_t (*cB)[40] = Bs[cur];

    // A 16x32 bf16 fragment: row = lane&15; halves interleave K in groups of 8
    v16bf af[2];
#pragma unroll
    for (int mm = 0; mm < 2; ++mm) {
      int rowA = wm * 32 + mm * 16 + ln15;
      v8bf a0 = *(const v8bf*)&cA[rowA][half * 8];
      v8bf a1 = *(const v8bf*)&cA[rowA][16 + half * 8];
      af[mm] = __builtin_shufflevector(a0, a1, 0, 1, 2, 3, 4, 5, 6, 7,
                                       8, 9, 10, 11, 12, 13, 14, 15);
    }
    // B 32x16 bf16 fragment: col = lane&15; halves interleave K in groups of 8
    v16bf bfv[2];
#pragma unroll
    for (int nn = 0; nn < 2; ++nn) {
      int colB = wn * 32 + nn * 16 + ln15;
      v8bf b0 = *(const v8bf*)&cB[colB][half * 8];
      v8bf b1 = *(const v8bf*)&cB[colB][16 + half * 8];
      bfv[nn] = __builtin_shufflevector(b0, b1, 0, 1, 2, 3, 4, 5, 6, 7,
                                        8, 9, 10, 11, 12, 13, 14, 15);
    }
#pragma unroll
    for (int mm = 0; mm < 2; ++mm)
#pragma unroll
      for (int nn = 0; nn < 2; ++nn)
        acc[mm][nn] = __builtin_amdgcn_wmma_f32_16x16x32_bf16(
            false, af[mm], false, bfv[nn], (short)0, acc[mm][nn], false, false);

    __syncthreads();   // protect LDS stage before it is refilled
  }

  // C/D layout: VGPR r -> M = r (lanes 0-15) / r+8 (lanes 16-31), N = lane&15
#pragma unroll
  for (int mm = 0; mm < 2; ++mm)
#pragma unroll
    for (int nn = 0; nn < 2; ++nn) {
      int gn = bN + wn * 32 + nn * 16 + ln15;
      if (gn < N) {
#pragma unroll
        for (int r = 0; r < 8; ++r) {
          int gm = bM + wm * 32 + mm * 16 + r + half * 8;
          float v = acc[mm][nn][r];
          if (bias) v += bias[gn];
          if (mode == 1) {                         // softplus -> bf16 (dt path)
            v = (v > 20.f) ? v : log1pf(__expf(v));
            Cb[(size_t)gm * N + gn] = (bf16_t)v;
          } else {
            Cf[(size_t)gm * N + gn] = v;
          }
        }
      }
    }
}

// ---------------------------------------------------------------------------
// Depthwise 4x4 conv (pad (1,2),(1,2)) + SiLU, fused scatter into the
// 4-direction scan tensor xsT: (K, M, DI) bf16 (WMMA A-operand layout).
// ---------------------------------------------------------------------------
__global__ void k_dwconv4_silu(const float* __restrict__ xz,
                               const float* __restrict__ cw,
                               const float* __restrict__ cb,
                               bf16_t* __restrict__ xsT)
{
  int i = blockIdx.x * blockDim.x + threadIdx.x;     // B*L*DI, d fastest
  if (i >= Bc * DIc * Lc) return;
  int d = i % DIc;
  int l = (i / DIc) % Lc;
  int b = i / (DIc * Lc);
  int h = l >> 6, w = l & 63;

  float s = cb[d];
#pragma unroll
  for (int ii = 0; ii < 4; ++ii) {
    int hh = h - 1 + ii;
    if ((unsigned)hh >= 64u) continue;
#pragma unroll
    for (int jj = 0; jj < 4; ++jj) {
      int ww = w - 1 + jj;
      if ((unsigned)ww >= 64u) continue;
      s += xz[((size_t)(b * Lc + (hh << 6) + ww)) * 768 + d] * cw[d * 16 + ii * 4 + jj];
    }
  }
  s = s / (1.f + __expf(-s));                        // SiLU
  bf16_t v = (bf16_t)s;

  size_t mb = (size_t)b * Lc;
  int lhw = (h << 6) + w;
  int lwh = (w << 6) + h;
  xsT[((size_t)0 * MT + mb + lhw) * DIc + d]            = v;
  xsT[((size_t)1 * MT + mb + lwh) * DIc + d]            = v;
  xsT[((size_t)2 * MT + mb + (Lc - 1 - lhw)) * DIc + d] = v;
  xsT[((size_t)3 * MT + mb + (Lc - 1 - lwh)) * DIc + d] = v;
}

// ---------------------------------------------------------------------------
// Selective scan: one thread per (b, d); loops k=0..3 internally so all four
// direction contributions (incl. Ds*u skip term and the flip/transpose
// re-indexing) accumulate race-free into ymg (B, L, DI).
// ---------------------------------------------------------------------------
__global__ void k_scan(const bf16_t* __restrict__ xsT,
                       const bf16_t* __restrict__ dtb,
                       const float*  __restrict__ xdbl,
                       const float*  __restrict__ A_log,
                       const float*  __restrict__ Dsp,
                       float* __restrict__ ymg)
{
  int t = blockIdx.x * blockDim.x + threadIdx.x;
  if (t >= Bc * DIc) return;
  int b = t / DIc, d = t % DIc;

  for (int k = 0; k < 4; ++k) {
    float Ak[16];
#pragma unroll
    for (int j = 0; j < 16; ++j)
      Ak[j] = -__expf(A_log[((k * DIc) + d) * 16 + j]);
    float Dk = Dsp[k * DIc + d];

    float h[16];
#pragma unroll
    for (int j = 0; j < 16; ++j) h[j] = 0.f;

    for (int l = 0; l < Lc; ++l) {
      size_t idx = (size_t)k * MT + (size_t)b * Lc + l;
      float u   = (float)xsT[idx * DIc + d];
      float dtv = (float)dtb[idx * DIc + d];
      const float* row = &xdbl[idx * 48];            // [0:12) dtr | [12:28) B | [28:44) C
      float du = dtv * u;
      float y  = 0.f;
#pragma unroll
      for (int j = 0; j < 16; ++j) {
        h[j] = h[j] * __expf(dtv * Ak[j]) + du * row[12 + j];
        y += h[j] * row[28 + j];
      }
      y += Dk * u;

      int l_std;
      if (k == 0)       l_std = l;
      else if (k == 1)  { int ww = l >> 6, hh = l & 63; l_std = (hh << 6) + ww; }
      else if (k == 2)  l_std = Lc - 1 - l;
      else              { int q = Lc - 1 - l; int ww = q >> 6, hh = q & 63; l_std = (hh << 6) + ww; }

      float* o = &ymg[((size_t)b * Lc + l_std) * DIc + d];
      if (k == 0) *o = y; else *o += y;
    }
  }
}

// ---------------------------------------------------------------------------
// LayerNorm over DI + SiLU(z) gating, emits bf16 rows for the out-proj GEMM.
// ---------------------------------------------------------------------------
__global__ __launch_bounds__(128) void k_ln_gate(
    const float* __restrict__ ymg, const float* __restrict__ xz,
    const float* __restrict__ g, const float* __restrict__ be,
    bf16_t* __restrict__ ybf)
{
  __shared__ float s1[128], s2[128];
  int m = blockIdx.x;
  const float* row = &ymg[(size_t)m * DIc];

  float a = 0.f, q = 0.f;
  for (int d = threadIdx.x; d < DIc; d += 128) { float v = row[d]; a += v; q += v * v; }
  s1[threadIdx.x] = a; s2[threadIdx.x] = q;
  __syncthreads();
  for (int s = 64; s > 0; s >>= 1) {
    if (threadIdx.x < s) { s1[threadIdx.x] += s1[threadIdx.x + s]; s2[threadIdx.x] += s2[threadIdx.x + s]; }
    __syncthreads();
  }
  float mu   = s1[0] / DIc;
  float var  = s2[0] / DIc - mu * mu;
  float rstd = rsqrtf(var + 1e-5f);

  for (int d = threadIdx.x; d < DIc; d += 128) {
    float v = (row[d] - mu) * rstd * g[d] + be[d];
    float z = xz[(size_t)m * 768 + DIc + d];
    v *= z / (1.f + __expf(-z));                     // * silu(z)
    ybf[(size_t)m * DIc + d] = (bf16_t)v;
  }
}

// ---------------------------------------------------------------------------
// Local branch: depthwise 3x3 + BN + ReLU, emits bf16 channel-last rows.
// ---------------------------------------------------------------------------
__global__ void k_dw3_bn_relu(const float* __restrict__ x,
                              const float* __restrict__ wgt, const float* __restrict__ bb,
                              const float* __restrict__ bg, const float* __restrict__ bbeta,
                              const float* __restrict__ bmean, const float* __restrict__ bvar,
                              bf16_t* __restrict__ lbf)
{
  int i = blockIdx.x * blockDim.x + threadIdx.x;     // B*L*C, c fastest
  if (i >= Bc * Cc * Lc) return;
  int c = i % Cc;
  int l = (i / Cc) % Lc;
  int b = i / (Cc * Lc);
  int h = l >> 6, w = l & 63;

  float s = bb[c];
#pragma unroll
  for (int ii = 0; ii < 3; ++ii) {
    int hh = h - 1 + ii;
    if ((unsigned)hh >= 64u) continue;
#pragma unroll
    for (int jj = 0; jj < 3; ++jj) {
      int ww = w - 1 + jj;
      if ((unsigned)ww >= 64u) continue;
      s += x[((size_t)(b * Cc + c)) * Lc + (hh << 6) + ww] * wgt[c * 9 + ii * 3 + jj];
    }
  }
  s = (s - bmean[c]) * rsqrtf(bvar[c] + 1e-5f) * bg[c] + bbeta[c];
  s = fmaxf(s, 0.f);
  lbf[((size_t)(b * Lc + l)) * Cc + c] = (bf16_t)s;
}

// ---------------------------------------------------------------------------
// Routing weights: spatial mean, tiny matvec, softmax over 2, final blend.
// ---------------------------------------------------------------------------
__global__ __launch_bounds__(256) void k_xmean(const float* __restrict__ x, float* __restrict__ xm) {
  __shared__ float sm[256];
  int bc = blockIdx.x;                               // b*C + c
  const float* p = &x[(size_t)bc * Lc];
  float a = 0.f;
  for (int l = threadIdx.x; l < Lc; l += 256) a += p[l];
  sm[threadIdx.x] = a;
  __syncthreads();
  for (int s = 128; s > 0; s >>= 1) {
    if (threadIdx.x < s) sm[threadIdx.x] += sm[threadIdx.x + s];
    __syncthreads();
  }
  if (threadIdx.x == 0) xm[bc] = sm[0] / Lc;
}

__global__ void k_wts(const float* __restrict__ xm, const float* __restrict__ rw,
                      const float* __restrict__ rb, float* __restrict__ wts) {
  __shared__ float lg[8];
  int t = threadIdx.x;
  if (t < 8) {
    int b = t >> 1, j = t & 1;
    float s = rb[j];
    for (int c = 0; c < Cc; ++c) s += xm[b * Cc + c] * rw[j * Cc + c];
    lg[t] = s;
  }
  __syncthreads();
  if (t < 8) {
    int b = t >> 1;
    float mx  = fmaxf(lg[b * 2], lg[b * 2 + 1]);
    float den = __expf(lg[b * 2] - mx) + __expf(lg[b * 2 + 1] - mx);
    wts[t] = __expf(lg[t] - mx) / den;
  }
}

__global__ void k_combine(const float* __restrict__ gout, const float* __restrict__ lout,
                          const float* __restrict__ wts, float* __restrict__ out) {
  int i = blockIdx.x * blockDim.x + threadIdx.x;     // output (B,C,H,W)
  if (i >= Bc * Cc * Lc) return;
  int l = i % Lc;
  int c = (i / Lc) % Cc;
  int b = i / (Cc * Lc);
  size_t m = (size_t)b * Lc + l;
  out[i] = wts[b * 2] * gout[m * Cc + c] + wts[b * 2 + 1] * lout[m * Cc + c];
}

// ---------------------------------------------------------------------------
extern "C" void kernel_launch(void* const* d_in, const int* in_sizes, int n_in,
                              void* d_out, int out_size, void* d_ws, size_t ws_size,
                              hipStream_t stream) {
  (void)in_sizes; (void)n_in; (void)out_size; (void)ws_size;

  const float* x         = (const float*)d_in[0];
  const float* W_in      = (const float*)d_in[1];
  const float* b_in      = (const float*)d_in[2];
  const float* conv_w    = (const float*)d_in[3];
  const float* conv_b    = (const float*)d_in[4];
  const float* x_proj_w  = (const float*)d_in[5];
  const float* dt_proj_w = (const float*)d_in[6];
  const float* dt_proj_b = (const float*)d_in[7];
  const float* A_log     = (const float*)d_in[8];
  const float* Dsp       = (const float*)d_in[9];
  const float* ln_g      = (const float*)d_in[10];
  const float* ln_b      = (const float*)d_in[11];
  const float* W_out     = (const float*)d_in[12];
  const float* b_out     = (const float*)d_in[13];
  const float* dw_w      = (const float*)d_in[14];
  const float* dw_b      = (const float*)d_in[15];
  const float* bn_g      = (const float*)d_in[16];
  const float* bn_b      = (const float*)d_in[17];
  const float* bn_mean   = (const float*)d_in[18];
  const float* bn_var    = (const float*)d_in[19];
  const float* pw_w      = (const float*)d_in[20];
  const float* pw_b      = (const float*)d_in[21];
  const float* r_w       = (const float*)d_in[22];
  const float* r_b       = (const float*)d_in[23];
  float* out = (float*)d_out;

  char* wsb = (char*)d_ws;
  size_t off = 0;
  auto take = [&](size_t bytes) -> void* {
    void* p = wsb + off;
    off += (bytes + 255) & ~(size_t)255;
    return p;
  };

  bf16_t* xbf    = (bf16_t*)take((size_t)MT * Cc * 2);
  bf16_t* WinT   = (bf16_t*)take((size_t)768 * Cc * 2);       // (768, 192)
  bf16_t* WoutT  = (bf16_t*)take((size_t)Cc * DIc * 2);       // (192, 384)
  bf16_t* pwBt   = (bf16_t*)take((size_t)Cc * Cc * 2);        // (192, 192) = pw_w as-is
  bf16_t* xpwBt  = (bf16_t*)take((size_t)4 * 64 * DIc * 2);   // (K, 64, 384)
  bf16_t* dtwBt  = (bf16_t*)take((size_t)4 * DIc * 32 * 2);   // (K, 384, 32)
  float*  xz     = (float*) take((size_t)MT * 768 * 4);
  bf16_t* xsT    = (bf16_t*)take((size_t)4 * MT * DIc * 2);
  float*  xdbl   = (float*) take((size_t)4 * MT * 48 * 4);
  bf16_t* dtrBf  = (bf16_t*)take((size_t)4 * MT * 32 * 2);
  bf16_t* dtBf   = (bf16_t*)take((size_t)4 * MT * DIc * 2);
  float*  ymg    = (float*) take((size_t)MT * DIc * 4);
  bf16_t* ybf    = (bf16_t*)take((size_t)MT * DIc * 2);
  float*  gout   = (float*) take((size_t)MT * Cc * 4);
  bf16_t* lbf    = (bf16_t*)take((size_t)MT * Cc * 2);
  float*  lout   = (float*) take((size_t)MT * Cc * 4);
  float*  xmean  = (float*) take((size_t)Bc * Cc * 4);
  float*  wtsb   = (float*) take(64);

  const int NBCL = Bc * Cc * Lc;                     // 3,145,728
  const int NBDL = Bc * DIc * Lc;                    // 6,291,456

  // --- weight / activation layout prep (Bt = B^T, K-contiguous rows) ---
  k_cvt_x      <<<(NBCL + 255) / 256, 256, 0, stream>>>(x, xbf);
  k_prep_WinT  <<<(768 * Cc + 255) / 256, 256, 0, stream>>>(W_in, WinT);
  k_prep_WoutT <<<(Cc * DIc + 255) / 256, 256, 0, stream>>>(W_out, WoutT);
  k_f32_to_bf16<<<(Cc * Cc + 255) / 256, 256, 0, stream>>>(pw_w, pwBt, Cc * Cc);
  k_prep_xpw   <<<(4 * 64 * DIc + 255) / 256, 256, 0, stream>>>(x_proj_w, xpwBt);
  k_prep_dtw   <<<(4 * DIc * 32 + 255) / 256, 256, 0, stream>>>(dt_proj_w, dtwBt);

  // --- in-projection: xz = xbf @ W_in + b_in   (M=16384, N=768, K=192) ---
  k_gemm_bf16<<<dim3(768 / 64, MT / 128), 256, 0, stream>>>(
      xbf, WinT, xz, nullptr, b_in, MT, 768, Cc, 0);

  // --- dwconv4 + SiLU fused into 4-direction scan tensor ---
  k_dwconv4_silu<<<(NBDL + 255) / 256, 256, 0, stream>>>(xz, conv_w, conv_b, xsT);

  // --- x_dbl[k] = xs[k] @ x_proj_w[k]^T  (N=48 logical, Bt padded to 64) ---
  for (int k = 0; k < 4; ++k)
    k_gemm_bf16<<<dim3(1, MT / 128), 256, 0, stream>>>(
        xsT + (size_t)k * MT * DIc, xpwBt + (size_t)k * 64 * DIc,
        xdbl + (size_t)k * MT * 48, nullptr, nullptr, MT, 48, DIc, 0);

  // --- dt = softplus(dtr @ dt_proj_w[k]^T + dt_proj_b[k])  (N=384, K=32) ---
  k_pack_dtr<<<(4 * MT * 32 + 255) / 256, 256, 0, stream>>>(xdbl, dtrBf);
  for (int k = 0; k < 4; ++k)
    k_gemm_bf16<<<dim3(DIc / 64, MT / 128), 256, 0, stream>>>(
        dtrBf + (size_t)k * MT * 32, dtwBt + (size_t)k * DIc * 32,
        nullptr, dtBf + (size_t)k * MT * DIc, dt_proj_b + k * DIc, MT, DIc, 32, 1);

  // --- selective scan, all 4 directions merged ---
  k_scan<<<(Bc * DIc + 255) / 256, 256, 0, stream>>>(xsT, dtBf, xdbl, A_log, Dsp, ymg);

  // --- LayerNorm + SiLU(z) gate -> bf16 rows ---
  k_ln_gate<<<MT, 128, 0, stream>>>(ymg, xz, ln_g, ln_b, ybf);

  // --- out-projection: gout = ybf @ W_out + b_out  (N=192, K=384) ---
  k_gemm_bf16<<<dim3(Cc / 64, MT / 128), 256, 0, stream>>>(
      ybf, WoutT, gout, nullptr, b_out, MT, Cc, DIc, 0);

  // --- local branch: dw3x3+BN+ReLU then pointwise GEMM (N=192, K=192) ---
  k_dw3_bn_relu<<<(NBCL + 255) / 256, 256, 0, stream>>>(
      x, dw_w, dw_b, bn_g, bn_b, bn_mean, bn_var, lbf);
  k_gemm_bf16<<<dim3(Cc / 64, MT / 128), 256, 0, stream>>>(
      lbf, pwBt, lout, nullptr, pw_b, MT, Cc, Cc, 0);

  // --- routing softmax + final blend ---
  k_xmean  <<<Bc * Cc, 256, 0, stream>>>(x, xmean);
  k_wts    <<<1, 64, 0, stream>>>(xmean, r_w, r_b, wtsb);
  k_combine<<<(NBCL + 255) / 256, 256, 0, stream>>>(gout, lout, wtsb, out);
}
